// CausalSelfAttention_87393994539323
// MI455X (gfx1250) — matrix operand order repaired and checked by reading, not compiled, not run on hardware
//
#include <hip/hip_runtime.h>
#include <hip/hip_bf16.h>

typedef __attribute__((ext_vector_type(16))) _Float16 v16h;
typedef __attribute__((ext_vector_type(8)))  _Float16 v8h;
typedef __attribute__((ext_vector_type(8)))  float    v8f;

#define BB 8
#define SS 1024
#define CC 768
#define HH 12
#define DD 64
#define MM (BB*SS)

// ---------------- Tensor Data Mover (guarded: compiles with or without) ----
#if defined(__has_builtin)
#if __has_builtin(__builtin_amdgcn_tensor_load_to_lds) && __has_builtin(__builtin_amdgcn_s_wait_tensorcnt)
#define HAVE_TDM 1
#endif
#endif
#ifndef HAVE_TDM
#define HAVE_TDM 0
#endif

#if HAVE_TDM
typedef unsigned int u32x4 __attribute__((ext_vector_type(4)));
typedef int          i32x8 __attribute__((ext_vector_type(8)));
typedef int          i32x4 __attribute__((ext_vector_type(4)));

// Generic 2D f16 tile load: tile_h rows x tile_w elems, row stride stride0
// elems, from a tensor advertised as dim0 x dim1 (from gsrc). D# per ISA ch.8.
__device__ __forceinline__ void tdm_load_2d(unsigned int lds_off, const void* gsrc,
                                            unsigned int tile_w, unsigned int tile_h,
                                            unsigned int stride0,
                                            unsigned int dim0, unsigned int dim1) {
    unsigned long long ga = (unsigned long long)(size_t)gsrc;
    u32x4 g0;
    g0.x = 1u;                                         // count=1 (valid user D#)
    g0.y = lds_off;                                    // lds_addr (bytes)
    g0.z = (unsigned int)ga;                           // global_addr[31:0]
    g0.w = (unsigned int)((ga >> 32) & 0x1ffffffu)     // global_addr[56:32]
         | (2u << 30);                                 // type = 2 ("image")
    i32x8 g1;
    g1[0] = (1 << 16);                                 // wg_mask=0, data_size=1 (2B)
    g1[1] = (int)(dim0 << 16);                         // tensor_dim0[15:0]
    g1[2] = (int)((dim0 >> 16) | (dim1 << 16));        // dim0[31:16] | dim1[15:0]
    g1[3] = (int)((dim1 >> 16) | (tile_w << 16));      // dim1[31:16] | tile_dim0
    g1[4] = (int)tile_h;                               // tile_dim1 | tile_dim2=0
    g1[5] = (int)stride0;                              // tensor_dim0_stride[31:0]
    g1[6] = 0;
    g1[7] = 0;
    i32x4 z4 = {0, 0, 0, 0};
#if __clang_major__ >= 23
    i32x8 z8 = {0, 0, 0, 0, 0, 0, 0, 0};
    __builtin_amdgcn_tensor_load_to_lds(g0, g1, z4, z4, z8, 0);
#else
    __builtin_amdgcn_tensor_load_to_lds(g0, g1, z4, z4, 0);
#endif
}
#define LDSOFF(p) ((unsigned int)(size_t)(const void*)(p))
#endif

// ---- WMMA helper: D = A(16x32 f16) * B(32x16 f16) + C(16x16 f32) ----
__device__ __forceinline__ v8f wmma16(v16h a, v16h b, v8f c) {
    return __builtin_amdgcn_wmma_f32_16x16x32_f16(
        false, a, false, b, (short)0, c, false, false);
}

// A-fragment (16x32, MxK): lane row = lane%16, h = lane/16.
__device__ __forceinline__ v16h load_a_f16(const _Float16* row, int k0, int h) {
    union { v16h v; v8h p[2]; } u;
    u.p[0] = *(const v8h*)(row + k0 + 16*h);
    u.p[1] = *(const v8h*)(row + k0 + 16*h + 8);
    return u.v;
}
// B-fragment (32x16, KxN) from [N][K] row-major: lane col = lane%16.
__device__ __forceinline__ v16h load_b_f16(const _Float16* row, int k0, int h) {
    union { v16h v; v8h p[2]; } u;
    const _Float16* p = row + k0 + 16*h;
    u.p[0] = *(const v8h*)(p);
    u.p[1] = *(const v8h*)(p + 8);
    return u.v;
}

// ---------------- Kernel 1: qkv = x @ w_qkv^T + b -> q/k/v [B,H,S,D] f16 ----
// 256 threads (8 waves), 128x64 tile, double-buffered LDS, register-staged
// global loads overlap WMMA.
__global__ __launch_bounds__(256)
void qkv_gemm(const float* __restrict__ x, const float* __restrict__ w,
              const float* __restrict__ bias,
              _Float16* __restrict__ qo, _Float16* __restrict__ ko,
              _Float16* __restrict__ vo) {
    __shared__ __attribute__((aligned(32))) _Float16 sA[2][128 * 32];
    __shared__ __attribute__((aligned(32))) _Float16 sB[2][64 * 32];

    const int lane = threadIdx.x & 31, l16 = lane & 15, hf = lane >> 4;
    const int wave = threadIdx.x >> 5;
    const int NB = (3 * CC) / 64;                    // 36 N-blocks
    const int bn = blockIdx.x % NB, bm = blockIdx.x / NB;
    const int m0 = bm * 128, n0 = bn * 64;

    const int ar = threadIdx.x >> 1, ah = threadIdx.x & 1;   // A: 128 rows x 2 halves
    const int br = threadIdx.x >> 2, bq = threadIdx.x & 3;   // B: 64 rows x 4 quarters
    const float* asrc = x + (size_t)(m0 + ar) * CC + ah * 16;
    const float* bsrc = w + (size_t)(n0 + br) * CC + bq * 8;

    float aReg[16], bReg[8];
#pragma unroll
    for (int i = 0; i < 16; ++i) aReg[i] = asrc[i];
#pragma unroll
    for (int i = 0; i < 8; ++i)  bReg[i] = bsrc[i];

    v8f acc0 = {}, acc1 = {}, acc2 = {}, acc3 = {};
    const int NK = CC / 32;                          // 24 K-steps
    for (int n = 0; n < NK; ++n) {
        const int cur = n & 1;
        __syncthreads();
        {
            _Float16* adst = sA[cur] + ar * 32 + ah * 16;
            _Float16* bdst = sB[cur] + br * 32 + bq * 8;
#pragma unroll
            for (int i = 0; i < 16; ++i) adst[i] = (_Float16)aReg[i];
#pragma unroll
            for (int i = 0; i < 8; ++i)  bdst[i] = (_Float16)bReg[i];
        }
        __syncthreads();
        if (n + 1 < NK) {                            // overlap next tile's loads
            const int k1 = (n + 1) * 32;
            __builtin_prefetch(asrc + k1 + 32, 0, 1);
            __builtin_prefetch(bsrc + k1 + 32, 0, 1);
#pragma unroll
            for (int i = 0; i < 16; ++i) aReg[i] = asrc[k1 + i];
#pragma unroll
            for (int i = 0; i < 8; ++i)  bReg[i] = bsrc[k1 + i];
        }
        const v16h a = load_a_f16(sA[cur] + (size_t)(wave * 16 + l16) * 32, 0, hf);
        v16h b;
        b = load_b_f16(sB[cur] + (size_t)( 0 + l16) * 32, 0, hf); acc0 = wmma16(a, b, acc0);
        b = load_b_f16(sB[cur] + (size_t)(16 + l16) * 32, 0, hf); acc1 = wmma16(a, b, acc1);
        b = load_b_f16(sB[cur] + (size_t)(32 + l16) * 32, 0, hf); acc2 = wmma16(a, b, acc2);
        b = load_b_f16(sB[cur] + (size_t)(48 + l16) * 32, 0, hf); acc3 = wmma16(a, b, acc3);
    }

    const v8f accs[4] = {acc0, acc1, acc2, acc3};
    const int mBase = m0 + wave * 16;
#pragma unroll
    for (int nt = 0; nt < 4; ++nt) {
        const int n = n0 + nt * 16 + l16;
        const float bn_ = bias[n];
        const int which = n / CC, cc = n % CC;
        const int hh = cc / DD, dd = cc % DD;
        _Float16* outp = (which == 0) ? qo : (which == 1) ? ko : vo;
#pragma unroll
        for (int r = 0; r < 8; ++r) {
            const int m = mBase + r + 8 * hf;
            const int bI = m / SS, s = m % SS;
            outp[(((size_t)bI * HH + hh) * SS + s) * DD + dd] =
                (_Float16)(accs[nt][r] + bn_);
        }
    }
}

// ---------------- Kernel 2: flash attention, double-buffered K/V tiles ----
__global__ __launch_bounds__(128)
void attn(const _Float16* __restrict__ q, const _Float16* __restrict__ k,
          const _Float16* __restrict__ v, const int* __restrict__ amask,
          _Float16* __restrict__ y) {
    __shared__ __attribute__((aligned(32))) _Float16 sK[2][32 * DD];  // [key][d]
    __shared__ __attribute__((aligned(32))) _Float16 sV[2][32 * DD];  // [key][d]
    __shared__ __attribute__((aligned(32))) _Float16 sP[4][16 * 32];

    const int lane = threadIdx.x & 31, l16 = lane & 15, hf = lane >> 4;
    const int wave = threadIdx.x >> 5;
    const int nrb = SS / 64;
    const int rb = blockIdx.x % nrb;
    const int bh = blockIdx.x / nrb;
    const int hh = bh % HH, bI = bh / HH;
    const int s0 = rb * 64 + wave * 16;              // first query row of wave

    const _Float16* qb = q + ((size_t)bI * HH + hh) * SS * DD;
    const _Float16* kb = k + ((size_t)bI * HH + hh) * SS * DD;
    const _Float16* vb = v + ((size_t)bI * HH + hh) * SS * DD;
    const int* mrow = amask + (size_t)bI * SS;

    const v16h aq0 = load_a_f16(qb + (size_t)(s0 + l16) * DD, 0, hf);
    const v16h aq1 = load_a_f16(qb + (size_t)(s0 + l16) * DD, 32, hf);

    float mrun[8], lrun[8];
    v8f o0 = {}, o1 = {}, o2 = {}, o3 = {};
#pragma unroll
    for (int r = 0; r < 8; ++r) { mrun[r] = -3.0e38f; lrun[r] = 0.0f; }

    _Float16* Pl = sP[wave];
    const int nTiles = (rb * 64 + 64) / 32;          // block-level causal tiles

#if HAVE_TDM
    if (threadIdx.x < 32) {                          // prologue: DMA tile 0
        tdm_load_2d(LDSOFF(sK[0]), kb, DD, 32, DD, DD, SS);
        tdm_load_2d(LDSOFF(sV[0]), vb, DD, 32, DD, DD, SS);
    }
#else
    const int strow = threadIdx.x >> 2, sq = threadIdx.x & 3;
    const _Float16* kst = kb + (size_t)strow * DD + sq * 16;
    const _Float16* vst = vb + (size_t)strow * DD + sq * 16;
    v8h kr0 = *(const v8h*)kst, kr1 = *(const v8h*)(kst + 8);
    v8h vr0 = *(const v8h*)vst, vr1 = *(const v8h*)(vst + 8);
#endif

    for (int n = 0; n < nTiles; ++n) {
        const int t0 = n * 32;
        const int cur = n & 1;
#if HAVE_TDM
        if (threadIdx.x < 32) __builtin_amdgcn_s_wait_tensorcnt(0);
        __syncthreads();                             // tile n visible; readers of
        if (threadIdx.x < 32 && n + 1 < nTiles) {    // buf^1 (iter n-1) are done
            tdm_load_2d(LDSOFF(sK[cur ^ 1]), kb + (size_t)(t0 + 32) * DD, DD, 32, DD, DD, SS);
            tdm_load_2d(LDSOFF(sV[cur ^ 1]), vb + (size_t)(t0 + 32) * DD, DD, 32, DD, DD, SS);
        }
#else
        __syncthreads();
        {
            _Float16* kd = sK[cur] + strow * DD + sq * 16;
            _Float16* vd = sV[cur] + strow * DD + sq * 16;
            *(v8h*)kd       = kr0;  *(v8h*)(kd + 8) = kr1;
            *(v8h*)vd       = vr0;  *(v8h*)(vd + 8) = vr1;
        }
        __syncthreads();
        if (n + 1 < nTiles) {                        // overlap next tile's loads
            kr0 = *(const v8h*)(kst + (size_t)(t0 + 32) * DD);
            kr1 = *(const v8h*)(kst + (size_t)(t0 + 32) * DD + 8);
            vr0 = *(const v8h*)(vst + (size_t)(t0 + 32) * DD);
            vr1 = *(const v8h*)(vst + (size_t)(t0 + 32) * DD + 8);
        }
#endif
        if (t0 < s0 + 16) {                          // wave-uniform causal skip
            const _Float16* sKc = sK[cur];
            const _Float16* sVc = sV[cur];
            const int keyA = t0 + l16;
            const int keyB = t0 + 16 + l16;
            v8f sAc = {}, sBc = {};
            v16h bk;
            bk = load_b_f16(sKc + (size_t)l16 * DD,        0,  hf); sAc = wmma16(aq0, bk, sAc);
            bk = load_b_f16(sKc + (size_t)l16 * DD,        32, hf); sAc = wmma16(aq1, bk, sAc);
            bk = load_b_f16(sKc + (size_t)(16 + l16) * DD, 0,  hf); sBc = wmma16(aq0, bk, sBc);
            bk = load_b_f16(sKc + (size_t)(16 + l16) * DD, 32, hf); sBc = wmma16(aq1, bk, sBc);

            const bool okA = (mrow[keyA] != 0);
            const bool okB = (mrow[keyB] != 0);
            float pA[8], pB[8];
#pragma unroll
            for (int r = 0; r < 8; ++r) {
                const int srow = s0 + r + 8 * hf;
                float vA = (okA && keyA <= srow) ? sAc[r] * 0.125f : -1.0e30f;
                float vB = (okB && keyB <= srow) ? sBc[r] * 0.125f : -1.0e30f;
                float mx = fmaxf(vA, vB);
#pragma unroll
                for (int off = 8; off >= 1; off >>= 1)
                    mx = fmaxf(mx, __shfl_xor(mx, off, 32));
                const float mnew  = fmaxf(mrun[r], mx);
                const float alpha = __expf(mrun[r] - mnew);
                const float eA = __expf(vA - mnew);
                const float eB = __expf(vB - mnew);
                float lsum = eA + eB;
#pragma unroll
                for (int off = 8; off >= 1; off >>= 1)
                    lsum += __shfl_xor(lsum, off, 32);
                lrun[r] = lrun[r] * alpha + lsum;
                mrun[r] = mnew;
                pA[r] = eA; pB[r] = eB;
                o0[r] *= alpha; o1[r] *= alpha; o2[r] *= alpha; o3[r] *= alpha;
            }
#pragma unroll
            for (int r = 0; r < 8; ++r) {            // C-layout -> A-layout for P
                const int row = r + 8 * hf;
                Pl[row * 32 + l16]      = (_Float16)pA[r];
                Pl[row * 32 + 16 + l16] = (_Float16)pB[r];
            }
            const v16h aP = load_a_f16(Pl + (size_t)l16 * 32, 0, hf);
#pragma unroll
            for (int c = 0; c < 4; ++c) {            // V B-frags gathered from [key][d]
                v16h bv;
#pragma unroll
                for (int e = 0; e < 16; ++e)
                    bv[e] = sVc[(16 * hf + e) * DD + (c * 16 + l16)];
                if      (c == 0) o0 = wmma16(aP, bv, o0);
                else if (c == 1) o1 = wmma16(aP, bv, o1);
                else if (c == 2) o2 = wmma16(aP, bv, o2);
                else             o3 = wmma16(aP, bv, o3);
            }
        }
    }
#pragma unroll
    for (int r = 0; r < 8; ++r) {                    // normalize, write y[B,S,C]
        const float inv = 1.0f / lrun[r];
        const int srow = s0 + r + 8 * hf;
        const size_t base = ((size_t)bI * SS + srow) * CC + hh * DD;
        y[base + 0 * 16 + l16] = (_Float16)(o0[r] * inv);
        y[base + 1 * 16 + l16] = (_Float16)(o1[r] * inv);
        y[base + 2 * 16 + l16] = (_Float16)(o2[r] * inv);
        y[base + 3 * 16 + l16] = (_Float16)(o3[r] * inv);
    }
}

// ---------------- Kernel 3: out = y @ w_out^T + b_out (fp32 output) ----
// A panel (f16, strided 2D tile) staged by TDM when available; B register-staged.
__global__ __launch_bounds__(256)
void out_gemm(const _Float16* __restrict__ y, const float* __restrict__ w,
              const float* __restrict__ bias, float* __restrict__ out) {
    __shared__ __attribute__((aligned(32))) _Float16 sA[2][128 * 32];
    __shared__ __attribute__((aligned(32))) _Float16 sB[2][64 * 32];

    const int lane = threadIdx.x & 31, l16 = lane & 15, hf = lane >> 4;
    const int wave = threadIdx.x >> 5;
    const int NB = CC / 64;                          // 12 N-blocks
    const int bn = blockIdx.x % NB, bm = blockIdx.x / NB;
    const int m0 = bm * 128, n0 = bn * 64;

    const int br = threadIdx.x >> 2, bq = threadIdx.x & 3;
    const float* bsrc = w + (size_t)(n0 + br) * CC + bq * 8;
    float bReg[8];
#pragma unroll
    for (int i = 0; i < 8; ++i) bReg[i] = bsrc[i];

    const _Float16* aTile = y + (size_t)m0 * CC;     // row m0, col k advances
#if HAVE_TDM
    if (threadIdx.x < 32)                            // prologue: DMA A tile 0
        tdm_load_2d(LDSOFF(sA[0]), aTile, 32, 128, CC, CC, MM);
#else
    const int ar = threadIdx.x >> 1, ah = threadIdx.x & 1;
    const _Float16* asrc = y + (size_t)(m0 + ar) * CC + ah * 16;
    v8h aReg0 = *(const v8h*)asrc, aReg1 = *(const v8h*)(asrc + 8);
#endif

    v8f acc0 = {}, acc1 = {}, acc2 = {}, acc3 = {};
    const int NK = CC / 32;                          // 24 K-steps
    for (int n = 0; n < NK; ++n) {
        const int cur = n & 1;
        const int k0 = n * 32;
#if HAVE_TDM
        if (threadIdx.x < 32) __builtin_amdgcn_s_wait_tensorcnt(0);
#endif
        __syncthreads();
        {
            _Float16* bdst = sB[cur] + br * 32 + bq * 8;
#pragma unroll
            for (int i = 0; i < 8; ++i) bdst[i] = (_Float16)bReg[i];
        }
#if HAVE_TDM
        if (threadIdx.x < 32 && n + 1 < NK)          // DMA next A tile under compute
            tdm_load_2d(LDSOFF(sA[cur ^ 1]), aTile + k0 + 32, 32, 128, CC, CC, MM);
#else
        {
            _Float16* adst = sA[cur] + ar * 32 + ah * 16;
            *(v8h*)adst = aReg0; *(v8h*)(adst + 8) = aReg1;
        }
#endif
        __syncthreads();
        if (n + 1 < NK) {                            // overlap next B loads
            __builtin_prefetch(bsrc + k0 + 64, 0, 1);
#pragma unroll
            for (int i = 0; i < 8; ++i) bReg[i] = bsrc[k0 + 32 + i];
#if !HAVE_TDM
            aReg0 = *(const v8h*)(asrc + k0 + 32);
            aReg1 = *(const v8h*)(asrc + k0 + 40);
#endif
        }
        const v16h a = load_a_f16(sA[cur] + (size_t)(wave * 16 + l16) * 32, 0, hf);
        v16h b;
        b = load_b_f16(sB[cur] + (size_t)( 0 + l16) * 32, 0, hf); acc0 = wmma16(a, b, acc0);
        b = load_b_f16(sB[cur] + (size_t)(16 + l16) * 32, 0, hf); acc1 = wmma16(a, b, acc1);
        b = load_b_f16(sB[cur] + (size_t)(32 + l16) * 32, 0, hf); acc2 = wmma16(a, b, acc2);
        b = load_b_f16(sB[cur] + (size_t)(48 + l16) * 32, 0, hf); acc3 = wmma16(a, b, acc3);
    }

    const v8f accs[4] = {acc0, acc1, acc2, acc3};
    const int mBase = m0 + wave * 16;
#pragma unroll
    for (int nt = 0; nt < 4; ++nt) {
        const int n = n0 + nt * 16 + l16;
        const float bn_ = bias[n];
#pragma unroll
        for (int r = 0; r < 8; ++r) {
            const int m = mBase + r + 8 * hf;
            out[(size_t)m * CC + n] = accs[nt][r] + bn_;
        }
    }
}

extern "C" void kernel_launch(void* const* d_in, const int* in_sizes, int n_in,
                              void* d_out, int out_size, void* d_ws, size_t ws_size,
                              hipStream_t stream) {
    const float* x      = (const float*)d_in[0];
    const float* w_qkv  = (const float*)d_in[1];
    const float* b_qkv  = (const float*)d_in[2];
    const float* w_out  = (const float*)d_in[3];
    const float* b_out  = (const float*)d_in[4];
    const int*   am     = (const int*)  d_in[5];
    float* out = (float*)d_out;

    const size_t QE = (size_t)BB * HH * SS * DD;     // q/k/v/y element counts
    _Float16* qw = (_Float16*)d_ws;
    _Float16* kw = qw + QE;
    _Float16* vw = kw + QE;
    _Float16* yw = vw + QE;

    {   // 1) QKV projection
        const int blocks = (MM / 128) * ((3 * CC) / 64);        // 2304
        qkv_gemm<<<blocks, 256, 0, stream>>>(x, w_qkv, b_qkv, qw, kw, vw);
    }
    {   // 2) flash attention
        const int blocks = BB * HH * (SS / 64);                 // 1536
        attn<<<blocks, 128, 0, stream>>>(qw, kw, vw, am, yw);
    }
    {   // 3) output projection
        const int blocks = (MM / 128) * (CC / 64);              // 768
        out_gemm<<<blocks, 256, 0, stream>>>(yw, w_out, b_out, out);
    }
}
